// SlotsScorer_71854802862137
// MI455X (gfx1250) — matrix operand order repaired and checked by reading, not compiled, not run on hardware
//
#include <hip/hip_runtime.h>
#include <cstdint>
#include <cstddef>

// ---------------------------------------------------------------------------
// Types for CDNA5 WMMA (gfx1250, wave32)
// ---------------------------------------------------------------------------
typedef __attribute__((ext_vector_type(8)))  __bf16 v8bf;
typedef __attribute__((ext_vector_type(16))) __bf16 v16bf;
typedef __attribute__((ext_vector_type(8)))  float  v8f;
typedef int v4i __attribute__((vector_size(16)));   // matches async-LDS builtin params

static constexpr int Hdim   = 1024;
static constexpr int Sdim   = 128;     // NUM_QUERIES
static constexpr int Bdim   = 512;
static constexpr int Adim   = 50;
static constexpr int NPAIR  = Bdim * Adim;        // 25600
static constexpr int KIN    = 3136;               // 3*1024 + 64
static constexpr int OUT_SCORES  = 0;                       // 512*51
static constexpr int OUT_COREF   = Bdim * (Adim + 1);       // 26112
static constexpr int OUT_CLUSTER = OUT_COREF + Sdim * NPAIR;// 3302912

// GEMM tiling: 256 threads = 8 waves as 4(M) x 2(N); each wave owns a 32x64
// C tile (2x4 WMMA 16x16 tiles). Block tile 128x128, K tile 64. LDS ~40KB.
static constexpr int BM = 128, BN = 128, BK = 64, PAD = 16; // (BK+PAD)*2B = 160B stride

// ---------------------------------------------------------------------------
// gfx1250 async global->LDS copies (ASYNCcnt-tracked, no VGPR staging)
// ---------------------------------------------------------------------------
#if defined(__has_builtin)
#if __has_builtin(__builtin_amdgcn_global_load_async_to_lds_b128) && \
    __has_builtin(__builtin_amdgcn_s_wait_asynccnt)
#define USE_ASYNC_LDS 1
#endif
#endif
#ifndef USE_ASYNC_LDS
#define USE_ASYNC_LDS 0
#endif

__device__ inline void cp16_g2l(const __bf16* __restrict__ g, __bf16* l) {
#if USE_ASYNC_LDS
  __builtin_amdgcn_global_load_async_to_lds_b128(
      (__attribute__((address_space(1))) v4i*)g,
      (__attribute__((address_space(3))) v4i*)l, 0, 0);
#else
  *(v8bf*)l = *(const v8bf*)g;
#endif
}
__device__ inline void async_fence() {
#if USE_ASYNC_LDS
  __builtin_amdgcn_s_wait_asynccnt(0);
#endif
}

// ---------------------------------------------------------------------------
// f32 -> bf16 conversion (grid-stride)
// ---------------------------------------------------------------------------
__global__ void f32_to_bf16_k(const float* __restrict__ s, __bf16* __restrict__ d, size_t n) {
  size_t i = (size_t)blockIdx.x * blockDim.x + threadIdx.x;
  size_t st = (size_t)gridDim.x * blockDim.x;
  for (; i < n; i += st) d[i] = (__bf16)s[i];
}

// Transpose f32 [K][N] -> bf16 [N][K] (one-time per launch; removes all
// per-tile in-LDS transposes from the GEMM inner loops). Dims % 32 == 0.
__global__ __launch_bounds__(256) void transpose_f32_to_bf16_k(
    const float* __restrict__ in, __bf16* __restrict__ out, int K, int N) {
  __shared__ float tile[32][33];
  const int tx = threadIdx.x & 31, ty = threadIdx.x >> 5;  // ty in 0..7
  const int k0 = blockIdx.x * 32, n0 = blockIdx.y * 32;
#pragma unroll
  for (int i = 0; i < 32; i += 8)
    tile[ty + i][tx] = in[(size_t)(k0 + ty + i) * N + n0 + tx];
  __syncthreads();
#pragma unroll
  for (int i = 0; i < 32; i += 8)
    out[(size_t)(n0 + ty + i) * K + k0 + tx] = (__bf16)tile[tx][ty + i];
}

// ---------------------------------------------------------------------------
// WMMA fragment loads (this code's contiguous k order, same for A and B).
// A (16-bit 16x32): lanes 0-15 K=0..15, lanes 16-31 K=16..31 (M = lane&15).
// B (16-bit 32x16): lanes 0-15 K=0..15, lanes 16-31 K=16..31 (N = lane&15).
// ---------------------------------------------------------------------------
__device__ inline v16bf load_frag_a(const __bf16* p) {
  v8bf lo = *(const v8bf*)p;
  v8bf hi = *(const v8bf*)(p + 8);
  return __builtin_shufflevector(lo, hi, 0,1,2,3,4,5,6,7,8,9,10,11,12,13,14,15);
}
__device__ inline v16bf load_frag_b(const __bf16* p) {
  v8bf lo = *(const v8bf*)p;
  v8bf hi = *(const v8bf*)(p + 8);
  return __builtin_shufflevector(lo, hi, 0,1,2,3,4,5,6,7,8,9,10,11,12,13,14,15);
}

template <int ACT, int OUT_BF16, int OUT_TRANS>
__device__ inline void gemm_epilogue(v8f acc[2][4], int m0, int n0, int wm, int wn,
                                     int lane, const float* bias, const float* resid,
                                     float scale, float* outF, __bf16* outB,
                                     int M, int N) {
  const int l15  = lane & 15;
  const int rsel = (lane >> 4) * 8;   // C layout: lanes 16-31 hold rows M+8..M+15
#pragma unroll
  for (int mi = 0; mi < 2; ++mi)
#pragma unroll
    for (int ni = 0; ni < 4; ++ni) {
      const int col = n0 + wn + ni * 16 + l15;
#pragma unroll
      for (int l = 0; l < 8; ++l) {
        const int row = m0 + wm + mi * 16 + rsel + l;
        float v = acc[mi][ni][l] * scale;
        if (bias)  v += bias[col];
        if (resid) v += resid[(size_t)row * N + col];
        if (ACT == 1) v = (v > 0.f) ? v : 0.01f * v;   // leaky_relu(0.01)
        if (ACT == 2) v = fmaxf(v, 0.f);               // relu
        const size_t o = OUT_TRANS ? ((size_t)col * M + row) : ((size_t)row * N + col);
        if (OUT_BF16) outB[o] = (__bf16)v; else outF[o] = v;
      }
    }
}

// ---------------------------------------------------------------------------
// Generic bf16 WMMA GEMM: C[M,N] = act(scale * A@B + bias + resid)
// A row-major [M][K]; B N-major [N][K] with row stride ldb (pre-transposed
// weights or naturally N-major activations). All dims are tile multiples.
// ---------------------------------------------------------------------------
template <int ACT, int OUT_BF16, int OUT_TRANS>
__global__ __launch_bounds__(256) void gemm_wmma_k(
    const __bf16* __restrict__ A, const __bf16* __restrict__ Bm,
    const float* __restrict__ bias, const float* __restrict__ resid, float scale,
    float* __restrict__ outF, __bf16* __restrict__ outB, int M, int N, int K, int ldb) {
  __shared__ __bf16 As[BM][BK + PAD];
  __shared__ __bf16 Bs[BN][BK + PAD];
  const int tid  = threadIdx.x;
  const int lane = tid & 31;
  const int wid  = tid >> 5;
  const int wm   = (wid & 3) * 32;
  const int wn   = (wid >> 2) * 64;
  const int m0   = blockIdx.x * BM;
  const int n0   = blockIdx.y * BN;
  v8f acc[2][4] = {};

  for (int k0 = 0; k0 < K; k0 += BK) {
    for (int g = tid; g < BM * BK / 8; g += 256) {           // A tile, 16B vectors
      const int r = g >> 3, kk = (g & 7) * 8;
      cp16_g2l(A + (size_t)(m0 + r) * K + k0 + kk, &As[r][kk]);
    }
    for (int g = tid; g < BN * BK / 8; g += 256) {           // B tile (N-major)
      const int n = g >> 3, kk = (g & 7) * 8;
      cp16_g2l(Bm + (size_t)(n0 + n) * ldb + k0 + kk, &Bs[n][kk]);
    }
    async_fence();
    __syncthreads();
    const int l15 = lane & 15;
    const int kloA = (lane < 16) ? 0 : 16;
    const int kloB = (lane < 16) ? 0 : 16;
#pragma unroll
    for (int ks = 0; ks < BK; ks += 32) {
      v16bf af[2], bf[4];
#pragma unroll
      for (int mi = 0; mi < 2; ++mi) af[mi] = load_frag_a(&As[wm + mi * 16 + l15][ks + kloA]);
#pragma unroll
      for (int ni = 0; ni < 4; ++ni) bf[ni] = load_frag_b(&Bs[wn + ni * 16 + l15][ks + kloB]);
#pragma unroll
      for (int mi = 0; mi < 2; ++mi)
#pragma unroll
        for (int ni = 0; ni < 4; ++ni)
          acc[mi][ni] = __builtin_amdgcn_wmma_f32_16x16x32_bf16(
              false, af[mi], false, bf[ni], (short)0, acc[mi][ni], false, false);
    }
    __syncthreads();
  }
  gemm_epilogue<ACT, OUT_BF16, OUT_TRANS>(acc, m0, n0, wm, wn, lane, bias, resid,
                                          scale, outF, outB, M, N);
}

// ---------------------------------------------------------------------------
// Fused pair-construction GEMM:
//   x = leaky_relu(concat(a, b, a*b, pw) @ hidden_w + hidden_b) -> bf16
// K segments [0,1024) mentions | [1024,2048) gathered | [2048,3072) product |
// [3072,3136) pw. Each 64-wide K tile lies in exactly one segment.
// B = hidden_w pre-transposed to N-major [1024][3136].
// ---------------------------------------------------------------------------
__global__ __launch_bounds__(256) void pair_gemm_k(
    const __bf16* __restrict__ ment, const __bf16* __restrict__ allm,
    const __bf16* __restrict__ pwb, const int* __restrict__ topidx,
    const __bf16* __restrict__ WT, const float* __restrict__ bias,
    __bf16* __restrict__ outB) {
  constexpr int M = NPAIR, N = Hdim, K = KIN;
  __shared__ __bf16 As[BM][BK + PAD];
  __shared__ __bf16 Bs[BN][BK + PAD];
  const int tid  = threadIdx.x;
  const int lane = tid & 31;
  const int wid  = tid >> 5;
  const int wm   = (wid & 3) * 32;
  const int wn   = (wid >> 2) * 64;
  const int m0   = blockIdx.x * BM;
  const int n0   = blockIdx.y * BN;
  v8f acc[2][4] = {};

  for (int k0 = 0; k0 < K; k0 += BK) {
    for (int g = tid; g < BM * BK / 8; g += 256) {
      const int r = g >> 3, kk = (g & 7) * 8;
      const int rg = m0 + r;
      const int k  = k0 + kk;
      if (k0 < 1024) {
        const int b = rg / Adim;
        cp16_g2l(ment + (size_t)b * Hdim + k, &As[r][kk]);
      } else if (k0 < 2048) {
        const int idx = topidx[rg];
        cp16_g2l(allm + (size_t)idx * Hdim + (k - 1024), &As[r][kk]);
      } else if (k0 < 3072) {
        const int b = rg / Adim;
        const int idx = topidx[rg];
        v8bf mv = *(const v8bf*)(ment + (size_t)b * Hdim + (k - 2048));
        v8bf gv = *(const v8bf*)(allm + (size_t)idx * Hdim + (k - 2048));
        v8bf val;
#pragma unroll
        for (int e = 0; e < 8; ++e) val[e] = (__bf16)((float)mv[e] * (float)gv[e]);
        *(v8bf*)&As[r][kk] = val;
      } else {
        cp16_g2l(pwb + (size_t)rg * 64 + (k - 3072), &As[r][kk]);
      }
    }
    for (int g = tid; g < BN * BK / 8; g += 256) {
      const int n = g >> 3, kk = (g & 7) * 8;
      cp16_g2l(WT + (size_t)(n0 + n) * K + k0 + kk, &Bs[n][kk]);
    }
    async_fence();
    __syncthreads();
    const int l15 = lane & 15;
    const int kloA = (lane < 16) ? 0 : 16;
    const int kloB = (lane < 16) ? 0 : 16;
#pragma unroll
    for (int ks = 0; ks < BK; ks += 32) {
      v16bf af[2], bf[4];
#pragma unroll
      for (int mi = 0; mi < 2; ++mi) af[mi] = load_frag_a(&As[wm + mi * 16 + l15][ks + kloA]);
#pragma unroll
      for (int ni = 0; ni < 4; ++ni) bf[ni] = load_frag_b(&Bs[wn + ni * 16 + l15][ks + kloB]);
#pragma unroll
      for (int mi = 0; mi < 2; ++mi)
#pragma unroll
        for (int ni = 0; ni < 4; ++ni)
          acc[mi][ni] = __builtin_amdgcn_wmma_f32_16x16x32_bf16(
              false, af[mi], false, bf[ni], (short)0, acc[mi][ni], false, false);
    }
    __syncthreads();
  }
  gemm_epilogue<1 /*leaky*/, 1 /*bf16*/, 0>(acc, m0, n0, wm, wn, lane, bias, nullptr,
                                            1.f, nullptr, outB, M, N);
}

// ---------------------------------------------------------------------------
// LayerNorm over C=1024 (one block per row; 4 elems/thread). Optional mask
// (row zeroed when !(score > -inf), matching jnp.where before LN).
// ---------------------------------------------------------------------------
template <int IN_F32>
__global__ __launch_bounds__(256) void layernorm_k(
    const void* __restrict__ inp, __bf16* __restrict__ outp,
    const float* __restrict__ gamma, const float* __restrict__ beta,
    const float* __restrict__ maskScores, int C) {
  __shared__ float red[256];
  const int row = blockIdx.x, tid = threadIdx.x;
  float mask = 1.f;
  if (maskScores && !(maskScores[row] > -INFINITY)) mask = 0.f;
  float x[4], s = 0.f;
#pragma unroll
  for (int t = 0; t < 4; ++t) {
    const int c = tid + t * 256;
    float v = IN_F32 ? ((const float*)inp)[(size_t)row * C + c]
                     : (float)((const __bf16*)inp)[(size_t)row * C + c];
    v *= mask; x[t] = v; s += v;
  }
  red[tid] = s; __syncthreads();
  for (int st = 128; st > 0; st >>= 1) { if (tid < st) red[tid] += red[tid + st]; __syncthreads(); }
  const float mean = red[0] / C; __syncthreads();
  float s2 = 0.f;
#pragma unroll
  for (int t = 0; t < 4; ++t) s2 += x[t] * x[t];
  red[tid] = s2; __syncthreads();
  for (int st = 128; st > 0; st >>= 1) { if (tid < st) red[tid] += red[tid + st]; __syncthreads(); }
  const float var = red[0] / C - mean * mean;
  const float rs  = rsqrtf(var + 1e-5f);
#pragma unroll
  for (int t = 0; t < 4; ++t) {
    const int c = tid + t * 256;
    outp[(size_t)row * C + c] = (__bf16)((x[t] - mean) * rs * gamma[c] + beta[c]);
  }
}

// ---------------------------------------------------------------------------
// Softmax over the 128 slots per input column j (axis=1 in reference).
// mode 0: attn = softmax + 1e-8 written in place (pre row-normalization)
// mode 1: coref_logits = min(softmax + 1e-8, 1) written to d_out
// ---------------------------------------------------------------------------
__global__ void softmax_slots_k(float* __restrict__ dots, float* __restrict__ outFinal, int mode) {
  const int NJ = NPAIR;
  const int j = blockIdx.x * 256 + threadIdx.x;
  if (j >= NJ) return;
  float m = -INFINITY;
  for (int i = 0; i < Sdim; ++i) m = fmaxf(m, dots[(size_t)i * NJ + j]);
  float s = 0.f;
  for (int i = 0; i < Sdim; ++i) s += expf(dots[(size_t)i * NJ + j] - m);
  const float inv = 1.f / s;
  for (int i = 0; i < Sdim; ++i) {
    float v = expf(dots[(size_t)i * NJ + j] - m) * inv + 1e-8f;
    if (mode) outFinal[(size_t)i * NJ + j] = fminf(v, 1.f);
    else      dots[(size_t)i * NJ + j] = v;
  }
}

__global__ void rowsum_k(const float* __restrict__ a, float* __restrict__ out, int Nj) {
  __shared__ float red[256];
  const int i = blockIdx.x;
  float s = 0.f;
  for (int j = threadIdx.x; j < Nj; j += 256) s += a[(size_t)i * Nj + j];
  red[threadIdx.x] = s; __syncthreads();
  for (int st = 128; st > 0; st >>= 1) { if (threadIdx.x < st) red[threadIdx.x] += red[threadIdx.x + st]; __syncthreads(); }
  if (!threadIdx.x) out[i] = red[0];
}

__global__ void norm_attn_k(const float* __restrict__ a, const float* __restrict__ rowsum,
                            __bf16* __restrict__ out, int Nj) {
  const size_t idx = (size_t)blockIdx.x * 256 + threadIdx.x;
  const int i = (int)(idx / Nj);
  out[idx] = (__bf16)(a[idx] / rowsum[i]);
}

// GRU cell combine (torch.nn.GRUCell semantics)
__global__ void gru_k(const float* __restrict__ gi, const float* __restrict__ gh,
                      const float* __restrict__ hprev, float* __restrict__ s2) {
  const int idx = blockIdx.x * 256 + threadIdx.x;      // 128*1024 total
  const int r = idx >> 10, h = idx & 1023;
  const size_t base = (size_t)r * 3072 + h;
  const float ir = gi[base], iz = gi[base + 1024], in_ = gi[base + 2048];
  const float hr = gh[base], hz = gh[base + 1024], hn = gh[base + 2048];
  const float rg = 1.f / (1.f + expf(-(ir + hr)));
  const float z  = 1.f / (1.f + expf(-(iz + hz)));
  const float n  = tanhf(in_ + rg * hn);
  s2[idx] = (1.f - z) * n + z * hprev[idx];
}

// cluster_logits = sigmoid(relu(s@cls_w1+b1) @ cls_w2 + b2), N=1 tail
__global__ void cls2_k(const float* __restrict__ cls1, const float* __restrict__ w2,
                       const float* __restrict__ b2, float* __restrict__ out) {
  __shared__ float red[256];
  const int i = blockIdx.x;
  float s = 0.f;
  for (int h = threadIdx.x; h < 512; h += 256) s += cls1[(size_t)i * 512 + h] * w2[h];
  red[threadIdx.x] = s; __syncthreads();
  for (int st = 128; st > 0; st >>= 1) { if (threadIdx.x < st) red[threadIdx.x] += red[threadIdx.x + st]; __syncthreads(); }
  if (!threadIdx.x) out[i] = 1.f / (1.f + expf(-(red[0] + b2[0])));
}

// scores = concat([eps, top_rough_scores], axis=1) -> [512, 51]
__global__ void scores_k(const float* __restrict__ rough, float* __restrict__ out) {
  const int idx = blockIdx.x * 256 + threadIdx.x;
  if (idx >= Bdim * (Adim + 1)) return;
  const int b = idx / (Adim + 1), c = idx % (Adim + 1);
  out[idx] = (c == 0) ? 1e-7f : rough[b * Adim + (c - 1)];
}

// ---------------------------------------------------------------------------
// Host orchestration
// ---------------------------------------------------------------------------
extern "C" void kernel_launch(void* const* d_in, const int* in_sizes, int n_in,
                              void* d_out, int out_size, void* d_ws, size_t ws_size,
                              hipStream_t stream) {
  if (n_in < 32) return;
  // inputs
  const float* all_m   = (const float*)d_in[0];
  const float* ment    = (const float*)d_in[1];
  const float* pw      = (const float*)d_in[2];
  const int*   topidx  = (const int*)  d_in[3];
  const float* rough   = (const float*)d_in[4];
  // params (dict insertion order)
  const float* hidden_w = (const float*)d_in[5];
  const float* hidden_b = (const float*)d_in[6];
  const float* query_e  = (const float*)d_in[7];
  const float* ln_in_g  = (const float*)d_in[8];
  const float* ln_in_b  = (const float*)d_in[9];
  const float* ln_sl_g  = (const float*)d_in[10];
  const float* ln_sl_b  = (const float*)d_in[11];
  const float* ln_ff_g  = (const float*)d_in[12];
  const float* ln_ff_b  = (const float*)d_in[13];
  const float* q_w = (const float*)d_in[14]; const float* q_b = (const float*)d_in[15];
  const float* k_w = (const float*)d_in[16]; const float* k_b = (const float*)d_in[17];
  const float* v_w = (const float*)d_in[18]; const float* v_b = (const float*)d_in[19];
  const float* gih_w = (const float*)d_in[20]; const float* ghh_w = (const float*)d_in[21];
  const float* gih_b = (const float*)d_in[22]; const float* ghh_b = (const float*)d_in[23];
  const float* m1_w = (const float*)d_in[24]; const float* m1_b = (const float*)d_in[25];
  const float* m2_w = (const float*)d_in[26]; const float* m2_b = (const float*)d_in[27];
  const float* c1_w = (const float*)d_in[28]; const float* c1_b = (const float*)d_in[29];
  const float* c2_w = (const float*)d_in[30]; const float* c2_b = (const float*)d_in[31];
  float* out = (float*)d_out;

  // workspace bump allocator (256B aligned)
  size_t off = 0;
  auto wsa = [&](size_t bytes) -> void* {
    void* p = (char*)d_ws + off;
    off += (bytes + 255) & ~(size_t)255;
    return p;
  };
  // weights, pre-transposed to N-major bf16 [N][K]
  __bf16* w_hidT = (__bf16*)wsa((size_t)KIN * Hdim * 2);     // [1024][3136]
  __bf16* w_qT   = (__bf16*)wsa((size_t)Hdim * Hdim * 2);    // [1024][1024]
  __bf16* w_kT   = (__bf16*)wsa((size_t)Hdim * Hdim * 2);
  __bf16* w_vT   = (__bf16*)wsa((size_t)Hdim * Hdim * 2);
  __bf16* w_gihT = (__bf16*)wsa((size_t)Hdim * 3072 * 2);    // [3072][1024]
  __bf16* w_ghhT = (__bf16*)wsa((size_t)Hdim * 3072 * 2);
  __bf16* w_m1T  = (__bf16*)wsa((size_t)Hdim * 2048 * 2);    // [2048][1024]
  __bf16* w_m2T  = (__bf16*)wsa((size_t)2048 * Hdim * 2);    // [1024][2048]
  __bf16* w_c1T  = (__bf16*)wsa((size_t)Hdim * 512 * 2);     // [512][1024]
  __bf16* ment_b = (__bf16*)wsa((size_t)Bdim * Hdim * 2);
  __bf16* allm_b = (__bf16*)wsa((size_t)2000 * Hdim * 2);
  __bf16* pw_b   = (__bf16*)wsa((size_t)NPAIR * 64 * 2);
  __bf16* x_b    = (__bf16*)wsa((size_t)NPAIR * Hdim * 2);   // hidden out, LN in-place
  __bf16* k_bf   = (__bf16*)wsa((size_t)NPAIR * Hdim * 2);   // [NPAIR][H] (N-major for dots)
  __bf16* v_bfT  = (__bf16*)wsa((size_t)NPAIR * Hdim * 2);   // [H][NPAIR] (N-major for updates)
  float*  dots   = (float*) wsa((size_t)Sdim * NPAIR * 4);   // reused as attn
  __bf16* attn_b = (__bf16*)wsa((size_t)Sdim * NPAIR * 2);
  float*  rowsum = (float*) wsa(Sdim * 4);
  float*  slots  = (float*) wsa((size_t)Sdim * Hdim * 4);
  __bf16* s_b    = (__bf16*)wsa((size_t)Sdim * Hdim * 2);
  __bf16* q_bf   = (__bf16*)wsa((size_t)Sdim * Hdim * 2);
  float*  upd    = (float*) wsa((size_t)Sdim * Hdim * 4);
  __bf16* upd_b  = (__bf16*)wsa((size_t)Sdim * Hdim * 2);
  __bf16* sprev_b= (__bf16*)wsa((size_t)Sdim * Hdim * 2);
  float*  gi     = (float*) wsa((size_t)Sdim * 3072 * 4);
  float*  gh     = (float*) wsa((size_t)Sdim * 3072 * 4);
  float*  s2     = (float*) wsa((size_t)Sdim * Hdim * 4);
  __bf16* ff_b   = (__bf16*)wsa((size_t)Sdim * Hdim * 2);
  __bf16* m1_bf  = (__bf16*)wsa((size_t)Sdim * 2048 * 2);
  float*  cls1   = (float*) wsa((size_t)Sdim * 512 * 4);
  (void)ws_size; (void)in_sizes; (void)out_size;

  auto cvt = [&](const float* s, __bf16* d, size_t n) {
    int blocks = (int)(((n + 255) / 256 < 4096) ? (n + 255) / 256 : 4096);
    f32_to_bf16_k<<<blocks, 256, 0, stream>>>(s, d, n);
  };
  auto tsp = [&](const float* s, __bf16* d, int K, int N) {  // [K][N] -> [N][K]
    transpose_f32_to_bf16_k<<<dim3(K / 32, N / 32), 256, 0, stream>>>(s, d, K, N);
  };
  tsp(hidden_w, w_hidT, KIN, Hdim);
  tsp(q_w, w_qT, Hdim, Hdim);
  tsp(k_w, w_kT, Hdim, Hdim);
  tsp(v_w, w_vT, Hdim, Hdim);
  tsp(gih_w, w_gihT, Hdim, 3072);
  tsp(ghh_w, w_ghhT, Hdim, 3072);
  tsp(m1_w, w_m1T, Hdim, 2048);
  tsp(m2_w, w_m2T, 2048, Hdim);
  tsp(c1_w, w_c1T, Hdim, 512);
  cvt(ment, ment_b, (size_t)Bdim * Hdim);
  cvt(all_m, allm_b, (size_t)2000 * Hdim);
  cvt(pw, pw_b, (size_t)NPAIR * 64);

  // slots <- query_embed
  (void)hipMemcpyAsync(slots, query_e, (size_t)Sdim * Hdim * 4,
                       hipMemcpyDeviceToDevice, stream);

  // x = leaky_relu(pair @ hidden_w + b)  (fused pair construction)
  pair_gemm_k<<<dim3(NPAIR / BM, Hdim / BN), 256, 0, stream>>>(
      ment_b, allm_b, pw_b, topidx, w_hidT, hidden_b, x_b);
  // inputs_n = LN(mask(x)) in place
  layernorm_k<0><<<NPAIR, 256, 0, stream>>>(x_b, x_b, ln_in_g, ln_in_b, rough, Hdim);
  // k (row-major out), v (transposed out: [H][NPAIR] for the attn@v GEMM)
  gemm_wmma_k<0,1,0><<<dim3(NPAIR/BM, Hdim/BN), 256, 0, stream>>>(
      x_b, w_kT, k_b, nullptr, 1.f, nullptr, k_bf, NPAIR, Hdim, Hdim, Hdim);
  gemm_wmma_k<0,1,1><<<dim3(NPAIR/BM, Hdim/BN), 256, 0, stream>>>(
      x_b, w_vT, v_b, nullptr, 1.f, nullptr, v_bfT, NPAIR, Hdim, Hdim, Hdim);

  const float scale = 0.03125f;  // 1024^-0.5
  for (int it = 0; it < 3; ++it) {
    layernorm_k<1><<<Sdim, 256, 0, stream>>>(slots, s_b, ln_sl_g, ln_sl_b, nullptr, Hdim);
    gemm_wmma_k<0,1,0><<<dim3(1, Hdim/BN), 256, 0, stream>>>(
        s_b, w_qT, q_b, nullptr, 1.f, nullptr, q_bf, Sdim, Hdim, Hdim, Hdim);
    gemm_wmma_k<0,0,0><<<dim3(1, NPAIR/BN), 256, 0, stream>>>(
        q_bf, k_bf, nullptr, nullptr, scale, dots, nullptr, Sdim, NPAIR, Hdim, Hdim);
    softmax_slots_k<<<NPAIR/256, 256, 0, stream>>>(dots, nullptr, 0);
    rowsum_k<<<Sdim, 256, 0, stream>>>(dots, rowsum, NPAIR);
    norm_attn_k<<<(Sdim*NPAIR)/256, 256, 0, stream>>>(dots, rowsum, attn_b, NPAIR);
    gemm_wmma_k<0,0,0><<<dim3(1, Hdim/BN), 256, 0, stream>>>(
        attn_b, v_bfT, nullptr, nullptr, 1.f, upd, nullptr, Sdim, Hdim, NPAIR, NPAIR);
    cvt(upd,   upd_b,   (size_t)Sdim * Hdim);
    cvt(slots, sprev_b, (size_t)Sdim * Hdim);
    gemm_wmma_k<0,0,0><<<dim3(1, 3072/BN), 256, 0, stream>>>(
        upd_b, w_gihT, gih_b, nullptr, 1.f, gi, nullptr, Sdim, 3072, Hdim, Hdim);
    gemm_wmma_k<0,0,0><<<dim3(1, 3072/BN), 256, 0, stream>>>(
        sprev_b, w_ghhT, ghh_b, nullptr, 1.f, gh, nullptr, Sdim, 3072, Hdim, Hdim);
    gru_k<<<(Sdim*Hdim)/256, 256, 0, stream>>>(gi, gh, slots, s2);
    layernorm_k<1><<<Sdim, 256, 0, stream>>>(s2, ff_b, ln_ff_g, ln_ff_b, nullptr, Hdim);
    gemm_wmma_k<2,1,0><<<dim3(1, 2048/BN), 256, 0, stream>>>(
        ff_b, w_m1T, m1_b, nullptr, 1.f, nullptr, m1_bf, Sdim, 2048, Hdim, Hdim);
    gemm_wmma_k<0,0,0><<<dim3(1, Hdim/BN), 256, 0, stream>>>(
        m1_bf, w_m2T, m2_b, s2 /*residual*/, 1.f, slots, nullptr, Sdim, Hdim, 2048, 2048);
  }

  // final attention + heads
  layernorm_k<1><<<Sdim, 256, 0, stream>>>(slots, s_b, ln_sl_g, ln_sl_b, nullptr, Hdim);
  gemm_wmma_k<0,1,0><<<dim3(1, Hdim/BN), 256, 0, stream>>>(
      s_b, w_qT, q_b, nullptr, 1.f, nullptr, q_bf, Sdim, Hdim, Hdim, Hdim);
  gemm_wmma_k<0,0,0><<<dim3(1, NPAIR/BN), 256, 0, stream>>>(
      q_bf, k_bf, nullptr, nullptr, scale, dots, nullptr, Sdim, NPAIR, Hdim, Hdim);
  softmax_slots_k<<<NPAIR/256, 256, 0, stream>>>(dots, out + OUT_COREF, 1);
  gemm_wmma_k<2,0,0><<<dim3(1, 512/BN), 256, 0, stream>>>(
      s_b, w_c1T, c1_b, nullptr, 1.f, cls1, nullptr, Sdim, 512, Hdim, Hdim);
  cls2_k<<<Sdim, 256, 0, stream>>>(cls1, c2_w, c2_b, out + OUT_CLUSTER);
  scores_k<<<(Bdim*(Adim+1) + 255)/256, 256, 0, stream>>>(rough, out + OUT_SCORES);
}
